// LGCN_22136261444117
// MI455X (gfx1250) — matrix operand order repaired and checked by reading, compiled, not run on hardware
//
#include <hip/hip_runtime.h>
#include <cmath>

#define EPS 1e-7f
#define N_NODES 50000
#define N_EDGES 400000

// Flip to 0 if the assembler rejects global_load_async_to_lds_b128
#define USE_ASYNC_COPY 1

typedef __bf16 bf16;
typedef __attribute__((ext_vector_type(16))) __bf16 v16bf;
typedef __attribute__((ext_vector_type(8)))  __bf16 v8bf;
typedef __attribute__((ext_vector_type(8)))  float  v8f;

union FragBF { v16bf v; v8bf h[2]; };

#if defined(__has_builtin)
#if __has_builtin(__builtin_amdgcn_sched_group_barrier)
#define SCHED_GROUP(mask, cnt) __builtin_amdgcn_sched_group_barrier((mask), (cnt), 0)
#endif
#endif
#ifndef SCHED_GROUP
#define SCHED_GROUP(mask, cnt)
#endif

#if USE_ASYNC_COPY
// byte offset of a generic pointer within the wave's LDS allocation
__device__ __forceinline__ uint32_t lds_byte_off(const void* p) {
  return (uint32_t)(uintptr_t)(__attribute__((address_space(3))) const void*)p;
}
// GLOBAL_LOAD_ASYNC_TO_LDS_B128: copies 16B from global vaddr to LDS offset vdst, tracked by ASYNCcnt
__device__ __forceinline__ void async_copy_b128(uint32_t ldsoff, const void* gsrc) {
  asm volatile("global_load_async_to_lds_b128 %0, %1, off"
               :: "v"(ldsoff), "v"(gsrc) : "memory");
}
__device__ __forceinline__ void wait_asynccnt0() {
  asm volatile("s_wait_asynccnt 0x0" ::: "memory");
}
#endif

// ---------------- block reduction (wave32) ----------------
__device__ __forceinline__ float block_reduce_sum(float v, float* smem) {
#pragma unroll
  for (int off = 16; off > 0; off >>= 1) v += __shfl_down(v, off);
  const int lane = threadIdx.x & 31;
  const int w    = threadIdx.x >> 5;
  if (lane == 0) smem[w] = v;
  __syncthreads();
  float r = (threadIdx.x < 8) ? smem[threadIdx.x] : 0.0f;
  if (w == 0) {
#pragma unroll
    for (int off = 4; off > 0; off >>= 1) r += __shfl_down(r, off);
    if (lane == 0) smem[0] = r;
  }
  __syncthreads();
  float out = smem[0];
  __syncthreads();
  return out;
}

// ---------------- weight convert: W (1023 x Din f32) -> Wb (1024x1024 bf16, zero padded)
__global__ void prep_w(const float* __restrict__ W, bf16* __restrict__ Wb, int Din) {
  int idx = blockIdx.x * 256 + threadIdx.x;     // over 1024*1024
  int o = idx >> 10, k = idx & 1023;
  float v = (o < 1023 && k < Din) ? W[(size_t)o * Din + k] : 0.0f;
  Wb[idx] = (bf16)v;
}

// ---------------- layer-1 fused: u = logmap0(proj(expmap0([0,x], K_IN)))[:,1:]  (K_IN = 1.0)
__global__ void u_first_kernel(const float* __restrict__ x, bf16* __restrict__ u) {
  const int n = blockIdx.x;
  const float* row = x + (size_t)n * 1024;
  __shared__ float smem[8];
  float s = 0.0f;
  for (int c = threadIdx.x; c < 1024; c += 256) { float v = row[c]; s += v * v; }
  float S  = block_reduce_sum(s, smem);
  float nx = sqrtf(S);
  float un = fmaxf(nx, EPS);                  // expmap0 norm clamp
  float t  = un;                              // un / sqrt(1.0)
  float sh = sinhf(t);
  float xsn = sh * (nx / un);                 // |xs| = sqrtK*sinh(t)*|x|/un, sqrtK=1
  float x0  = sqrtf(1.0f + xsn * xsn);        // proj time component
  float theta = fmaxf(x0, 1.0f + EPS);        // logmap0 clamp
  float d  = acoshf(theta);
  float xn = fmaxf(xsn, EPS);
  float coef = (d / xn) * (sh / un);          // u = d * xs/xn ; xs = sinh(t)*x/un
  bf16* uo = u + (size_t)n * 1024;
  for (int c = threadIdx.x; c < 1024; c += 256) uo[c] = (bf16)(coef * row[c]);
}

// ---------------- general u = logmap0(X, k)[:,1:]; X rows: [x0, xs(1023)], stride 1024; pad col 1023 with 0
__global__ void u_gen_kernel(const float* __restrict__ X, bf16* __restrict__ u, float sqrtk) {
  const int n = blockIdx.x;
  const float* row = X + (size_t)n * 1024;
  __shared__ float smem[8];
  float s = 0.0f;
  for (int c = 1 + threadIdx.x; c < 1024; c += 256) { float v = row[c]; s += v * v; }
  float S  = block_reduce_sum(s, smem);
  float xn = fmaxf(sqrtf(S), EPS);
  float theta = fmaxf(row[0] / sqrtk, 1.0f + EPS);
  float d  = sqrtk * acoshf(theta);
  float coef = d / xn;
  bf16* uo = u + (size_t)n * 1024;
  for (int c = threadIdx.x; c < 1024; c += 256) {
    float v = (c < 1023) ? coef * row[1 + c] : 0.0f;
    uo[c] = (bf16)v;
  }
}

// ---------------- WMMA GEMM with LDS double-buffered staging ----------------
// out[m, 1+o] = sum_k A[m,k]*B[o,k] + bias[o]
// A: M x 1024 bf16 row-major, B: 1024 x 1024 bf16 row-major (padded W).
// Block: 256 thr = 8 waves; block tile 128 M x 64 N; wave -> 16 M x 64 N; K-step 32.
// Next tile staged global->LDS with GLOBAL_LOAD_ASYNC_TO_LDS_B128 (ASYNCcnt, no VGPRs);
// fragments: all 10 ds_load_b128 issued before the 4 back-to-back WMMAs (partial dscnt waits);
// s_wait_asynccnt 0 just before the barrier.
#define AROW 20   // dwords per staged A/B row (16 data + 4 pad)
__global__ __launch_bounds__(256, 1)
void gemm_kernel(const bf16* __restrict__ A, const bf16* __restrict__ B,
                 const float* __restrict__ bias, float* __restrict__ out, int M) {
  __shared__ __align__(16) uint32_t As[2][128 * AROW];   // 2 x 10 KB
  __shared__ __align__(16) uint32_t Bs[2][64 * AROW];    // 2 x  5 KB

  const int t    = threadIdx.x;
  const int lane = t & 31;
  const int wave = t >> 5;
  const int half = lane >> 4;
  const int l16  = lane & 15;
  const int n0 = blockIdx.x * 64;        // column tile = fast grid dim (A reuse in L2)
  const int m0 = blockIdx.y * 128;       // row tile
  const int mw = m0 + wave * 16;         // this wave's 16 rows

  // per-thread staging: one 16B chunk (8 bf16) per tile-row-group
  const int srow = t >> 2;               // 0..63
  const int schk = (t & 3) * 8;          // k offset of chunk within 32
  int ar0 = m0 + srow;      if (ar0 >= M) ar0 = M - 1;
  int ar1 = m0 + srow + 64; if (ar1 >= M) ar1 = M - 1;
  const bf16* agp0 = A + (size_t)ar0 * 1024 + schk;
  const bf16* agp1 = A + (size_t)ar1 * 1024 + schk;
  const bf16* bgp  = B + (size_t)(n0 + srow) * 1024 + schk;
  uint32_t* asl0 = &As[0][0] + srow * AROW + (t & 3) * 4;
  uint32_t* asl1 = asl0 + 64 * AROW;
  uint32_t* bsl  = &Bs[0][0] + srow * AROW + (t & 3) * 4;
  const int STRA = 128 * AROW, STRB = 64 * AROW;       // dwords between buffers
#if USE_ASYNC_COPY
  const uint32_t aoff0 = lds_byte_off(asl0);
  const uint32_t aoff1 = lds_byte_off(asl1);
  const uint32_t boff  = lds_byte_off(bsl);
#endif

  // prologue: stage K-step 0 into buffer 0 (prefetch the next A stripe meanwhile)
  __builtin_prefetch(agp0 + 32, 0, 1);
  __builtin_prefetch(agp1 + 32, 0, 1);
  *(uint4*)asl0 = *(const uint4*)(agp0);
  *(uint4*)asl1 = *(const uint4*)(agp1);
  *(uint4*)bsl  = *(const uint4*)(bgp);
  __syncthreads();

  v8f acc[4] = {};
  for (int ks = 0; ks < 32; ++ks) {
    const int buf = ks & 1;
    const int nb  = buf ^ 1;
    // next tile K offset, clamped (last iteration redundantly re-stages tile 31)
    int kn = (ks + 1) * 32; if (kn > 992) kn = 992;

#if USE_ASYNC_COPY
    // DMA next tile straight into the other LDS buffer (no VGPR round-trip)
    async_copy_b128(aoff0 + (uint32_t)nb * (STRA * 4), agp0 + kn);
    async_copy_b128(aoff1 + (uint32_t)nb * (STRA * 4), agp1 + kn);
    async_copy_b128(boff  + (uint32_t)nb * (STRB * 4), bgp  + kn);
#else
    uint4 ga0 = *(const uint4*)(agp0 + kn);
    uint4 ga1 = *(const uint4*)(agp1 + kn);
    uint4 gb0 = *(const uint4*)(bgp  + kn);
#endif

    // fragment loads from current LDS buffer (all issued before the wmma group)
    const uint32_t* arow = &As[buf][(wave * 16 + l16) * AROW];
    FragBF a;
    a.h[0] = *(const v8bf*)(arow + half * 4);       // A 16x32: K = half*8 .. +7
    a.h[1] = *(const v8bf*)(arow + 8 + half * 4);   //          K = 16+half*8 .. +7
    FragBF bfr[4];
#pragma unroll
    for (int tt = 0; tt < 4; ++tt) {
      const uint32_t* p = &Bs[buf][(tt * 16 + l16) * AROW + half * 4];  // K = half*8..+7, 16+half*8..+7
      bfr[tt].h[0] = *(const v8bf*)(p);
      bfr[tt].h[1] = *(const v8bf*)(p + 8);
    }
#pragma unroll
    for (int tt = 0; tt < 4; ++tt)
      acc[tt] = __builtin_amdgcn_wmma_f32_16x16x32_bf16(
          false, a.v, false, bfr[tt].v, (short)0, acc[tt], false, false);

#if !USE_ASYNC_COPY
    *(uint4*)(asl0 + nb * STRA) = ga0;
    *(uint4*)(asl1 + nb * STRA) = ga1;
    *(uint4*)(bsl  + nb * STRB) = gb0;
#endif

    SCHED_GROUP(0x100, 10);   // all fragment DS reads first (in order)
    SCHED_GROUP(0x008, 4);    // then 4x wmma (partial dscnt waits between)

#if USE_ASYNC_COPY
    wait_asynccnt0();         // next tile resident in LDS before the barrier
#endif
    __syncthreads();
  }

#pragma unroll
  for (int tt = 0; tt < 4; ++tt) {
    int col = n0 + tt * 16 + l16;
    float bv = (col < 1023) ? bias[col] : 0.0f;
#pragma unroll
    for (int r = 0; r < 8; ++r) {
      int m = mw + r + half * 8;                    // C/D layout: M = r + 8*(lane/16)
      if (m < M && col < 1023)
        out[(size_t)m * 1024 + 1 + col] = acc[tt][r] + bv;
    }
  }
}

// ---------------- row-wise expmap0([0, mx], k) in place: H rows stride 1024, spatial at cols 1..1023
__global__ void row_exp_kernel(float* __restrict__ H, float sqrtk) {
  const int n = blockIdx.x;
  float* row = H + (size_t)n * 1024;
  __shared__ float smem[8];
  float s = 0.0f;
  for (int c = 1 + threadIdx.x; c < 1024; c += 256) { float v = row[c]; s += v * v; }
  float S  = block_reduce_sum(s, smem);
  float un = fmaxf(sqrtf(S), EPS);
  float t  = un / sqrtk;
  float scale = sqrtk * sinhf(t) / un;
  for (int c = 1 + threadIdx.x; c < 1024; c += 256) row[c] *= scale;
  if (threadIdx.x == 0) row[0] = sqrtk * coshf(t);
}

// ---------------- zero f32 buffer (N*1024 floats as float4)
__global__ void zero_kernel(float4* __restrict__ p) {
  p[(size_t)blockIdx.x * 256 + threadIdx.x] = float4{0.f, 0.f, 0.f, 0.f};
}

// ---------------- edge scatter-add: one wave per edge
__global__ void edge_kernel(const float* __restrict__ H, const int* __restrict__ src,
                            const int* __restrict__ dst, const float* __restrict__ ew,
                            float* __restrict__ Macc, int nE) {
  const int e = blockIdx.x * 8 + (threadIdx.x >> 5);
  if (e >= nE) return;
  const int sN = src[e], dN = dst[e];
  const float w = ew[e];
  const float* hrow = H + (size_t)sN * 1024;
  float* mrow = Macc + (size_t)dN * 1024;
  const int lane = threadIdx.x & 31;
  for (int c = lane; c < 1024; c += 32)
    unsafeAtomicAdd(&mrow[c], w * hrow[c]);   // global_atomic_add_f32
}

// ---------------- fused centroid -> relu(logmap0(.,kin)) -> proj(expmap0(.,kout))
__global__ void post_kernel(const float* __restrict__ Min, float* __restrict__ out,
                            float sqrtkin, float sqrtkout, float kout) {
  const int n = blockIdx.x;
  const float* row = Min + (size_t)n * 1024;
  __shared__ float smem[8];
  float s2 = 0.0f, p2 = 0.0f;
  for (int c = 1 + threadIdx.x; c < 1024; c += 256) {
    float v = row[c]; s2 += v * v;
    float p = fmaxf(v, 0.0f); p2 += p * p;
  }
  float S2 = block_reduce_sum(s2, smem);
  float P2 = block_reduce_sum(p2, smem);
  float m0 = row[0];
  float l_inner = S2 - m0 * m0;                       // Lorentz inner (spatial - time)
  float denom = sqrtf(fmaxf(fabsf(l_inner), EPS));
  float sc = sqrtkin / denom;                         // centroid scale (>0)
  float h0 = sc * m0;
  float hn = fmaxf(sc * sqrtf(S2), EPS);              // logmap0 spatial-norm clamp
  float theta = fmaxf(h0 / sqrtkin, 1.0f + EPS);
  float d  = sqrtkin * acoshf(theta);
  float f  = d / hn;                                  // >= 0, so relu keeps sign of h_s
  float un = fmaxf(f * sc * sqrtf(P2), EPS);          // |relu(t_s)|
  float tt = un / sqrtkout;
  float g  = sqrtkout * sinhf(tt) / un;
  float coef = g * f * sc;
  float* orow = out + (size_t)n * 1024;
  for (int c = 1 + threadIdx.x; c < 1024; c += 256)
    orow[c] = coef * fmaxf(row[c], 0.0f);
  if (threadIdx.x == 0)
    orow[0] = sqrtf(kout + coef * coef * P2);         // proj time component
}

extern "C" void kernel_launch(void* const* d_in, const int* in_sizes, int n_in,
                              void* d_out, int out_size, void* d_ws, size_t ws_size,
                              hipStream_t stream) {
  (void)in_sizes; (void)n_in; (void)out_size; (void)ws_size;
  const float* x    = (const float*)d_in[0];
  const int*   esrc = (const int*)  d_in[1];
  const int*   edst = (const int*)  d_in[2];
  const float* ew   = (const float*)d_in[3];
  const float* W1   = (const float*)d_in[4];
  const float* b1   = (const float*)d_in[5];
  const float* W2   = (const float*)d_in[6];
  const float* b2   = (const float*)d_in[7];
  float* out = (float*)d_out;

  const float K_HID = 1.1f, K_OUT = 1.2f;
  const float SQ_IN  = 1.0f;
  const float SQ_HID = sqrtf(K_HID);
  const float SQ_OUT = sqrtf(K_OUT);

  // workspace carve-up
  char* ws = (char*)d_ws;
  const size_t WB_BYTES = (size_t)1024 * 1024 * 2;          // 2 MB each
  const size_t U_BYTES  = (size_t)N_NODES * 1024 * 2;       // 100 MB
  const size_t F_BYTES  = (size_t)N_NODES * 1024 * 4;       // 200 MB
  bf16*  wb1 = (bf16*)(ws);
  bf16*  wb2 = (bf16*)(ws + WB_BYTES);
  bf16*  uA  = (bf16*)(ws + 2 * WB_BYTES);
  float* hB  = (float*)(ws + 2 * WB_BYTES + U_BYTES);
  float* mC  = (float*)(ws + 2 * WB_BYTES + U_BYTES + F_BYTES);

  prep_w<<<4096, 256, 0, stream>>>(W1, wb1, 1024);
  prep_w<<<4096, 256, 0, stream>>>(W2, wb2, 1023);

  dim3 gg(16, (N_NODES + 127) / 128);   // x = column tiles (fast, shares A stripe in L2)

  // ---- layer 1 (k_in = 1.0, k_out = 1.1) ----
  u_first_kernel<<<N_NODES, 256, 0, stream>>>(x, uA);
  gemm_kernel<<<gg, 256, 0, stream>>>(uA, wb1, b1, hB, N_NODES);
  row_exp_kernel<<<N_NODES, 256, 0, stream>>>(hB, SQ_IN);
  zero_kernel<<<N_NODES, 256, 0, stream>>>((float4*)mC);
  edge_kernel<<<(N_EDGES + 7) / 8, 256, 0, stream>>>(hB, esrc, edst, ew, mC, N_EDGES);
  post_kernel<<<N_NODES, 256, 0, stream>>>(mC, hB, SQ_IN, SQ_HID, K_HID);

  // ---- layer 2 (k_in = 1.1, k_out = 1.2) ----
  u_gen_kernel<<<N_NODES, 256, 0, stream>>>(hB, uA, SQ_HID);
  gemm_kernel<<<gg, 256, 0, stream>>>(uA, wb2, b2, mC, N_NODES);
  row_exp_kernel<<<N_NODES, 256, 0, stream>>>(mC, SQ_HID);
  zero_kernel<<<N_NODES, 256, 0, stream>>>((float4*)hB);
  edge_kernel<<<(N_EDGES + 7) / 8, 256, 0, stream>>>(mC, esrc, edst, ew, hB, N_EDGES);
  post_kernel<<<N_NODES, 256, 0, stream>>>(hB, out, SQ_HID, SQ_OUT, K_OUT);
}
